// RNNModel_39161511805589
// MI455X (gfx1250) — matrix operand order repaired and checked by reading, not compiled, not run on hardware
//
#include <hip/hip_runtime.h>
#include <hip/hip_bf16.h>
#include <math.h>

typedef __attribute__((ext_vector_type(2)))  float  v2f;
typedef __attribute__((ext_vector_type(4)))  float  v4f;
typedef __attribute__((ext_vector_type(8)))  float  v8f;
typedef __attribute__((ext_vector_type(8)))  __bf16 v8bf;
typedef __attribute__((ext_vector_type(16))) __bf16 v16bf;
typedef __attribute__((ext_vector_type(4)))  unsigned int u32x4;
typedef __attribute__((ext_vector_type(8)))  int i32x8;
typedef __attribute__((ext_vector_type(4)))  int i32x4;

#define KC 32
#define LDPAD 2
#define LDST (KC + LDPAD)     // f32 LDS row stride (34 floats)

#define BKC 32                // bf16 GEMM K-chunk (one WMMA K-step)
#define BLD 40                // bf16 LDS row stride (80 B): bank-spread, 16B-aligned rows

#define TDM_A_BYTES (64 * BLD * 2)                      // 5120
#define TDM_B_BYTES (128 * BLD * 2)                     // 10240
#define TDM_STAGE_BYTES (2 * TDM_A_BYTES + 2 * TDM_B_BYTES)  // 30720 per buffer

// -------------------------------------------------------------------------
// fp32 -> bf16 hi/lo split (RNE), a ~= hi + lo
// -------------------------------------------------------------------------
__device__ __forceinline__ unsigned short f32_to_bf16_rne(float f) {
    unsigned u = __float_as_uint(f);
    unsigned lsb = (u >> 16) & 1u;
    u += 0x7fffu + lsb;
    return (unsigned short)(u >> 16);
}
__device__ __forceinline__ float bf16_to_f32(unsigned short h) {
    return __uint_as_float((unsigned)h << 16);
}

__global__ void split_f32_bf16(const float* __restrict__ src,
                               unsigned short* __restrict__ hi,
                               unsigned short* __restrict__ lo, int n) {
    int i = blockIdx.x * blockDim.x + threadIdx.x;
    if (i >= n) return;
    float v = src[i];
    unsigned short h = f32_to_bf16_rne(v);
    hi[i] = h;
    lo[i] = f32_to_bf16_rne(v - bf16_to_f32(h));
}

// -------------------------------------------------------------------------
// Embedding gather: X[row, h] = W_inp[h, tok[row]] + b_inp[h]
// -------------------------------------------------------------------------
__global__ void embed_gather(const int* __restrict__ tok,
                             const float* __restrict__ W_inp,
                             const float* __restrict__ b_inp,
                             float* __restrict__ X, int TB, int H, int V) {
    int idx = blockIdx.x * blockDim.x + threadIdx.x;
    if (idx >= TB * H) return;
    int row = idx / H;
    int h   = idx - row * H;
    int t   = tok[row];
    X[idx]  = W_inp[(size_t)h * V + t] + b_inp[h];
}

__global__ void copyf(const float* __restrict__ src, float* __restrict__ dst, int n) {
    int i = blockIdx.x * blockDim.x + threadIdx.x;
    if (i < n) dst[i] = src[i];
}

// -------------------------------------------------------------------------
// Exact fp32 WMMA GEMM: C[M,N] = A[M,K]*B[N,K]^T + bias0 + bias1
// Block 128 thr (4 waves), 64x64 tile, wave -> 32x32 quadrant.
// -------------------------------------------------------------------------
__global__ __launch_bounds__(128)
void gemm_nt_wmma_f32(const float* __restrict__ A, const float* __restrict__ B,
                      const float* __restrict__ bias0, const float* __restrict__ bias1,
                      float* __restrict__ C, int M, int N, int K) {
    __shared__ float lA[64 * LDST];
    __shared__ float lB[64 * LDST];

    const int tid  = threadIdx.x;
    const int wave = tid >> 5;
    const int lane = tid & 31;

    const int m0 = blockIdx.y * 64;
    const int n0 = blockIdx.x * 64;

    const int wm = (wave & 1) * 32;
    const int wn = (wave >> 1) * 32;

    v8f acc[2][2] = {};

    const int lm = lane & 15;
    const int kb = (lane >> 4) * 2;

    const int lrow = tid >> 3;
    const int lcol = (tid & 7) * 4;

    for (int k0 = 0; k0 < K; k0 += KC) {
        // batch all global loads into registers first, then store to LDS
        v4f ra[4], rb[4];
#pragma unroll
        for (int r = 0; r < 4; ++r) {
            ra[r] = *(const v4f*)(A + (size_t)(m0 + lrow + 16 * r) * K + k0 + lcol);
            rb[r] = *(const v4f*)(B + (size_t)(n0 + lrow + 16 * r) * K + k0 + lcol);
        }
#pragma unroll
        for (int r = 0; r < 4; ++r) {
            *(v4f*)(lA + (lrow + 16 * r) * LDST + lcol) = ra[r];
            *(v4f*)(lB + (lrow + 16 * r) * LDST + lcol) = rb[r];
        }
        __syncthreads();

#pragma unroll
        for (int k = 0; k < KC; k += 4) {
            v2f a[2], b[2];
#pragma unroll
            for (int i = 0; i < 2; ++i) {
                a[i] = *(const v2f*)(lA + (wm + i * 16 + lm) * LDST + k + kb);
                b[i] = *(const v2f*)(lB + (wn + i * 16 + lm) * LDST + k + kb);
            }
#pragma unroll
            for (int i = 0; i < 2; ++i)
#pragma unroll
                for (int j = 0; j < 2; ++j)
                    acc[i][j] = __builtin_amdgcn_wmma_f32_16x16x4_f32(
                        false, a[i], false, b[j], (short)0, acc[i][j], false, false);
        }
        __syncthreads();
    }

    const int cn = lane & 15;
    const int cm = (lane >> 4) * 8;
#pragma unroll
    for (int j = 0; j < 2; ++j) {
        const int gn = n0 + wn + j * 16 + cn;
        float bsum = 0.f;
        if (bias0) bsum += bias0[gn];
        if (bias1) bsum += bias1[gn];
#pragma unroll
        for (int i = 0; i < 2; ++i)
#pragma unroll
            for (int r = 0; r < 8; ++r)
                C[(size_t)(m0 + wm + i * 16 + cm + r) * N + gn] = acc[i][j][r] + bsum;
    }
}

// -------------------------------------------------------------------------
// TDM descriptor issue: DMA a [nrows x BKC] bf16 tile (row stride K elements)
// from global into LDS at lds_off, padding each 64B row to 80B (BLD shorts).
// D# layout per CDNA5 ISA ch.8 (group0 128b, group1 256b; groups 2/3 zero).
// -------------------------------------------------------------------------
__device__ __forceinline__ void tdm_load_tile(unsigned lds_off,
                                              const unsigned short* gbase,
                                              unsigned row0, unsigned k0,
                                              unsigned K, unsigned nrows,
                                              unsigned tensor_rows) {
    unsigned long long ga =
        (unsigned long long)(size_t)(gbase + (size_t)row0 * K + k0);
    u32x4 g0;
    g0[0] = 1u;                                    // count=1 (valid), user mode
    g0[1] = lds_off;                               // lds_addr (bytes)
    g0[2] = (unsigned)(ga & 0xffffffffull);        // global_addr[31:0]
    g0[3] = (unsigned)((ga >> 32) & 0x1ffffffull)  // global_addr[56:32]
          | (2u << 30);                            // type=2 ("image")

    i32x8 g1;
    g1[0] = (int)((1u << 16)      // data_size = 2 bytes
                | (1u << 20)      // pad_enable
                | (3u << 22)      // pad_interval: 16 DWORDs (64B) between pads
                | (3u << 25));    // pad_amount:  4 DWORDs (16B) pad
    g1[1] = (int)((K & 0xffffu) << 16);                              // tensor_dim0 lo
    g1[2] = (int)(((K >> 16) & 0xffffu) | ((tensor_rows & 0xffffu) << 16)); // dim0 hi | dim1 lo
    g1[3] = (int)(((tensor_rows >> 16) & 0xffffu) | ((unsigned)BKC << 16)); // dim1 hi | tile_dim0
    g1[4] = (int)nrows;            // tile_dim1 (tile_dim2 = 0)
    g1[5] = (int)K;                // tensor_dim0_stride[31:0]
    g1[6] = 0;                     // stride hi | dim1_stride lo (unused, 2D)
    g1[7] = 0;

    i32x4 z4 = {0, 0, 0, 0};
    i32x8 z8 = {0, 0, 0, 0, 0, 0, 0, 0};
    // 6-arg toolchain variant: (g0, g1, g2, g3, extra, cpol)
    __builtin_amdgcn_tensor_load_to_lds(g0, g1, z4, z4, z8, 0);
}

// -------------------------------------------------------------------------
// Split-bf16 WMMA GEMM (decoder): C ~= Ah*Bh + Ah*Bl + Al*Bh + bias (f32 acc)
// Block 128 thr (4 waves); block tile 64x128; wave tile 32x64.
// Tiles staged by TDM DMA, double-buffered LDS, overlapped with WMMA.
// NOTE: all LDS is dynamic so TDM lds_addr offsets start at 0.
// -------------------------------------------------------------------------
__device__ __forceinline__ v16bf frag_a(const unsigned short* rowp, int kg) {
    const unsigned short* p = rowp + kg * 8;
    v8bf x = *(const v8bf*)(p);
    v8bf y = *(const v8bf*)(p + 16);
    return __builtin_shufflevector(x, y, 0, 1, 2, 3, 4, 5, 6, 7,
                                         8, 9, 10, 11, 12, 13, 14, 15);
}
__device__ __forceinline__ v16bf frag_b(const unsigned short* rowp, int kg) {
    const unsigned short* p = rowp + kg * 8;
    v8bf x = *(const v8bf*)(p);
    v8bf y = *(const v8bf*)(p + 16);
    return __builtin_shufflevector(x, y, 0, 1, 2, 3, 4, 5, 6, 7,
                                         8, 9, 10, 11, 12, 13, 14, 15);
}

__global__ __launch_bounds__(128)
void gemm_nt_wmma_bf16x2_tdm(const unsigned short* __restrict__ Ah,
                             const unsigned short* __restrict__ Al,
                             const unsigned short* __restrict__ Bh,
                             const unsigned short* __restrict__ Bl,
                             const float* __restrict__ bias, float* __restrict__ C,
                             int M, int N, int K) {
    extern __shared__ char smem[];   // 2 * TDM_STAGE_BYTES

    const int tid  = threadIdx.x;
    const int wave = tid >> 5;
    const int lane = tid & 31;

    const int m0 = blockIdx.y * 64;
    const int n0 = blockIdx.x * 128;

    const int wm = (wave & 1) * 32;   // wave M offset in tile
    const int wn = (wave >> 1) * 64;  // wave N offset in tile

    const int lm = lane & 15;
    const int kg = lane >> 4;         // K-group (ISA 7.12.2 16-bit layouts)

    v8f acc[2][4] = {};

    const int nchunk = K / BKC;

    auto issue = [&](int buf, int k0) {
        unsigned base = (unsigned)(buf * TDM_STAGE_BYTES);
        tdm_load_tile(base,                               Ah, (unsigned)m0, (unsigned)k0,
                      (unsigned)K, 64u, (unsigned)M);
        tdm_load_tile(base + TDM_A_BYTES,                 Al, (unsigned)m0, (unsigned)k0,
                      (unsigned)K, 64u, (unsigned)M);
        tdm_load_tile(base + 2 * TDM_A_BYTES,             Bh, (unsigned)n0, (unsigned)k0,
                      (unsigned)K, 128u, (unsigned)N);
        tdm_load_tile(base + 2 * TDM_A_BYTES + TDM_B_BYTES, Bl, (unsigned)n0, (unsigned)k0,
                      (unsigned)K, 128u, (unsigned)N);
    };

    if (wave == 0) issue(0, 0);

    for (int c = 0; c < nchunk; ++c) {
        if (wave == 0) {
            if (c + 1 < nchunk) {
                issue((c + 1) & 1, (c + 1) * BKC);            // prefetch next chunk
                __builtin_amdgcn_s_wait_tensorcnt(4);         // current chunk's 4 done
            } else {
                __builtin_amdgcn_s_wait_tensorcnt(0);
            }
        }
        __syncthreads();

        const unsigned short* sAh =
            (const unsigned short*)(smem + (c & 1) * TDM_STAGE_BYTES);
        const unsigned short* sAl = sAh + 64 * BLD;
        const unsigned short* sBh = sAl + 64 * BLD;
        const unsigned short* sBl = sBh + 128 * BLD;

        v16bf ah[2], al[2];
#pragma unroll
        for (int i = 0; i < 2; ++i) {
            ah[i] = frag_a(sAh + (wm + i * 16 + lm) * BLD, kg);
            al[i] = frag_a(sAl + (wm + i * 16 + lm) * BLD, kg);
        }
#pragma unroll
        for (int j = 0; j < 4; ++j) {
            v16bf bh = frag_b(sBh + (wn + j * 16 + lm) * BLD, kg);
            v16bf bl = frag_b(sBl + (wn + j * 16 + lm) * BLD, kg);
#pragma unroll
            for (int i = 0; i < 2; ++i) {
                acc[i][j] = __builtin_amdgcn_wmma_f32_16x16x32_bf16(
                    false, ah[i], false, bh, (short)0, acc[i][j], false, false);
                acc[i][j] = __builtin_amdgcn_wmma_f32_16x16x32_bf16(
                    false, ah[i], false, bl, (short)0, acc[i][j], false, false);
                acc[i][j] = __builtin_amdgcn_wmma_f32_16x16x32_bf16(
                    false, al[i], false, bh, (short)0, acc[i][j], false, false);
            }
        }
        __syncthreads();
    }

    const int cn = lane & 15;
    const int cm = kg * 8;
#pragma unroll
    for (int j = 0; j < 4; ++j) {
        const int gn = n0 + wn + j * 16 + cn;
        const float bs = bias ? bias[gn] : 0.f;
#pragma unroll
        for (int i = 0; i < 2; ++i)
#pragma unroll
            for (int r = 0; r < 8; ++r)
                C[(size_t)(m0 + wm + i * 16 + cm + r) * N + gn] = acc[i][j][r] + bs;
    }
}

// -------------------------------------------------------------------------
// One RNN timestep (exact fp32): Hout = tanh(XWt + Hin * W_hh^T), M=32.
// -------------------------------------------------------------------------
__global__ __launch_bounds__(64)
void rnn_step(const float* __restrict__ Hin, const float* __restrict__ W_hh,
              const float* __restrict__ XWt, float* __restrict__ Hout,
              float* __restrict__ Ot, int H) {
    const int lane = threadIdx.x & 31;
    const int wave = threadIdx.x >> 5;
    const int n0   = (blockIdx.x * 2 + wave) * 16;

    const int lm = lane & 15;
    const int kb = (lane >> 4) * 2;

    v8f acc0 = {}, acc1 = {};
    for (int k = 0; k < H; k += 4) {
        v2f a0 = *(const v2f*)(Hin + (size_t)lm * H + k + kb);
        v2f a1 = *(const v2f*)(Hin + (size_t)(16 + lm) * H + k + kb);
        v2f b  = *(const v2f*)(W_hh + (size_t)(n0 + lm) * H + k + kb);
        acc0 = __builtin_amdgcn_wmma_f32_16x16x4_f32(false, a0, false, b, (short)0, acc0, false, false);
        acc1 = __builtin_amdgcn_wmma_f32_16x16x4_f32(false, a1, false, b, (short)0, acc1, false, false);
    }

    const int cn = lane & 15;
    const int cm = (lane >> 4) * 8;
#pragma unroll
    for (int r = 0; r < 8; ++r) {
        int m0 = cm + r;
        float v0 = tanhf(acc0[r] + XWt[(size_t)m0 * H + n0 + cn]);
        Hout[(size_t)m0 * H + n0 + cn] = v0;
        Ot[(size_t)m0 * H + n0 + cn]   = v0;
        int m1 = 16 + cm + r;
        float v1 = tanhf(acc1[r] + XWt[(size_t)m1 * H + n0 + cn]);
        Hout[(size_t)m1 * H + n0 + cn] = v1;
        Ot[(size_t)m1 * H + n0 + cn]   = v1;
    }
}

// -------------------------------------------------------------------------
extern "C" void kernel_launch(void* const* d_in, const int* in_sizes, int n_in,
                              void* d_out, int out_size, void* d_ws, size_t ws_size,
                              hipStream_t stream) {
    const int T = 128, B = 32, V = 32000, H = 512;
    const int TB = T * B;
    const size_t TBH = (size_t)TB * H;
    const size_t BH  = (size_t)B * H;
    const size_t VH  = (size_t)V * H;

    const int*   tok    = (const int*)d_in[0];
    const float* hidden = (const float*)d_in[1];
    const float* W_inp  = (const float*)d_in[2];
    const float* b_inp  = (const float*)d_in[3];
    const float* W_ih   = (const float*)d_in[4];
    const float* b_ih   = (const float*)d_in[5];
    const float* W_hh   = (const float*)d_in[6];
    const float* b_hh   = (const float*)d_in[7];
    const float* W_dec  = (const float*)d_in[8];
    const float* b_dec  = (const float*)d_in[9];

    float* out   = (float*)d_out;                 // decoded [T*B, V]
    float* out_h = out + (size_t)TB * V;          // h_last [B*H]

    float* X  = (float*)d_ws;                     // [TB, H]
    float* XW = X + TBH;                          // [TB, H]
    float* O  = XW + TBH;                         // [TB, H]
    float* H0 = O + TBH;                          // [B, H]
    float* H1 = H0 + BH;                          // [B, H]
    unsigned short* Ohi = (unsigned short*)(H1 + BH);
    unsigned short* Olo = Ohi + TBH;
    unsigned short* Wdh = Olo + TBH;
    unsigned short* Wdl = Wdh + VH;

    const size_t need_base = (3 * TBH + 2 * BH) * sizeof(float);
    const size_t need_bf16 = need_base + (2 * TBH + 2 * VH) * sizeof(unsigned short);
    const bool use_bf16 = (ws_size >= need_bf16);

    // 1) encoder == embedding gather
    embed_gather<<<(int)((TBH + 255) / 256), 256, 0, stream>>>(tok, W_inp, b_inp, X, TB, H, V);

    // 2) hoisted input projection: XW = X @ W_ih^T + (b_ih + b_hh)   (exact fp32)
    {
        dim3 grid(H / 64, TB / 64);
        gemm_nt_wmma_f32<<<grid, 128, 0, stream>>>(X, W_ih, b_ih, b_hh, XW, TB, H, H);
    }

    // 3) h0 = hidden[0]
    copyf<<<(int)((BH + 255) / 256), 256, 0, stream>>>(hidden, H0, (int)BH);

    // 4) sequential recurrence (exact fp32, error-feedback path)
    float* hin = H0;
    float* hout = H1;
    for (int t = 0; t < T; ++t) {
        rnn_step<<<H / 32, 64, 0, stream>>>(hin, W_hh,
                                            XW + (size_t)t * B * H, hout,
                                            O + (size_t)t * B * H, H);
        float* tmp = hin; hin = hout; hout = tmp;
    }
    copyf<<<(int)((BH + 255) / 256), 256, 0, stream>>>(hin, out_h, (int)BH);

    // 5) decoder: out = O @ W_dec^T + b_dec  (134 GFLOP -> split-bf16 WMMA + TDM)
    if (use_bf16) {
        split_f32_bf16<<<(int)((TBH + 255) / 256), 256, 0, stream>>>(O, Ohi, Olo, (int)TBH);
        split_f32_bf16<<<(int)((VH + 255) / 256), 256, 0, stream>>>(W_dec, Wdh, Wdl, (int)VH);
        dim3 grid(V / 128, TB / 64);
        gemm_nt_wmma_bf16x2_tdm<<<grid, 128, 2 * TDM_STAGE_BYTES, stream>>>(
            Ohi, Olo, Wdh, Wdl, b_dec, out, TB, V, H);
    } else {
        dim3 grid(V / 64, TB / 64);
        gemm_nt_wmma_f32<<<grid, 128, 0, stream>>>(O, W_dec, b_dec, nullptr, out, TB, V, H);
    }
}
